// WindowAttention_17540646437652
// MI455X (gfx1250) — matrix operand order repaired and checked by reading, not compiled, not run on hardware
//
#include <hip/hip_runtime.h>
#include <hip/hip_bf16.h>

typedef __bf16 bf16;
typedef __attribute__((ext_vector_type(16))) __bf16 v16bf;
typedef __attribute__((ext_vector_type(8)))  __bf16 v8bf;
typedef __attribute__((ext_vector_type(8)))  float  v8f;

#define DIM   384
#define NHEAD 12
#define NTOK  49
#define NWIN  4096

// ---- CDNA5 WMMA 16-bit fragment address helpers (CDNA5 ISA) ----
__device__ __forceinline__ int b_off(int ko, int ni) {   // B 32x16 (KxN)
  int lane = ni + (((ko >> 3) & 1) ? 16 : 0);
  return lane * 16 + (ko & 7) + ((ko >= 16) ? 8 : 0);
}

__device__ __forceinline__ v8f wmma_bf16(v16bf a, v16bf b, v8f c) {
  return __builtin_amdgcn_wmma_f32_16x16x32_bf16(false, a, false, b, (short)0, c, false, false);
}

// ---------------- prep kernels: pack weights into B-fragment layout ----------------
__global__ __launch_bounds__(256) void pack_wqkv(const float* __restrict__ w, bf16* __restrict__ dst) {
  int idx = blockIdx.x * 256 + threadIdx.x;            // over 1152*384
  if (idx >= DIM * 3 * DIM) return;
  int n = idx / DIM;
  int k = idx % DIM;
  dst[(((k >> 5) * 72 + (n >> 4)) << 9) + b_off(k & 31, n & 15)] = (bf16)w[n * DIM + k];
}

__global__ __launch_bounds__(256) void pack_wproj(const float* __restrict__ w, bf16* __restrict__ dst) {
  int idx = blockIdx.x * 256 + threadIdx.x;            // over 384*384
  if (idx >= DIM * DIM) return;
  int n = idx / DIM;
  int k = idx % DIM;
  dst[(((k >> 5) * 24 + (n >> 4)) << 9) + b_off(k & 31, n & 15)] = (bf16)w[n * DIM + k];
}

__global__ __launch_bounds__(256) void pack_bias(const float* __restrict__ rpb, const int* __restrict__ rel,
                                                 float* __restrict__ dst) {
  int idx = blockIdx.x * 256 + threadIdx.x;            // over 12*49*49
  if (idx >= NHEAD * NTOK * NTOK) return;
  int h  = idx / (NTOK * NTOK);
  int mn = idx % (NTOK * NTOK);
  dst[h * (NTOK * NTOK) + mn] = rpb[rel[mn] * NHEAD + h];
}

// ---------------- device helpers for the fused kernel ----------------

// One QKV chunk: 3 of the head's 6 column tiles (c=0: q0,q1,k0 ; c=1: k1,v0,v1)
__device__ __forceinline__ void qkv_chunk(
    const float* __restrict__ xw, const float* __restrict__ qkv_b,
    const bf16* __restrict__ wqkv,
    bf16* stQ, bf16* stK, bf16* stV,
    int head, int c, int lane, int lc, int half, float scale)
{
  const v8f vzero = {0.f, 0.f, 0.f, 0.f, 0.f, 0.f, 0.f, 0.f};
  v8f acc[3][4];
  #pragma unroll
  for (int t = 0; t < 3; ++t)
    #pragma unroll
    for (int mt = 0; mt < 4; ++mt) acc[t][mt] = vzero;

  #pragma unroll 1
  for (int kt = 0; kt < 12; ++kt) {
    // A fragments of x (fp32 -> bf16 on the fly); clamp row + select-0 to keep loads unconditional
    v16bf af[4];
    const int kb = kt * 32 + ((lane >= 16) ? 8 : 0);
    #pragma unroll
    for (int mt = 0; mt < 4; ++mt) {
      const int mi = lc + mt * 16;
      const bool ok = (mi < NTOK);
      const int mcl = ok ? mi : (NTOK - 1);
      const float* xr = xw + mcl * DIM + kb;
      #pragma unroll
      for (int j = 0; j < 8; ++j) {
        const int k = ((j >= 4) ? 16 : 0) + (j & 3) * 2;
        float2 p2 = *(const float2*)(xr + k);
        af[mt][2 * j]     = (bf16)(ok ? p2.x : 0.f);
        af[mt][2 * j + 1] = (bf16)(ok ? p2.y : 0.f);
      }
    }
    #pragma unroll
    for (int t = 0; t < 3; ++t) {
      const int idx6 = c * 3 + t;                                // 0..5
      const int tile = (idx6 >> 1) * 24 + 2 * head + (idx6 & 1); // global col tile 0..71
      v16bf bfr = *(const v16bf*)(wqkv + (((kt * 72 + tile) << 9) + lane * 16));
      #pragma unroll
      for (int mt = 0; mt < 4; ++mt)
        acc[t][mt] = wmma_bf16(af[mt], bfr, acc[t][mt]);
    }
  }

  // ---- write Q/K/V into fragment-packed staging ----
  #pragma unroll
  for (int t = 0; t < 3; ++t) {
    const int idx6 = c * 3 + t;
    const int sect = idx6 >> 1;                        // 0=q 1=k 2=v
    const int nf   = (idx6 & 1) * 16 + lc;             // 0..31 within head
    const float bv = qkv_b[sect * DIM + head * 32 + nf];
    if (sect == 0) {
      // Q: A-packed; dest = mt*512 + (r+half + 16*((nf&15)>=8))*16 + jpart
      const int jpart = (((nf >= 16) ? 4 : 0) + ((nf & 7) >> 1)) * 2 + (nf & 1);
      const int base0 = (half + (((nf & 15) >= 8) ? 16 : 0)) * 16 + jpart;
      #pragma unroll
      for (int mt = 0; mt < 4; ++mt)
        #pragma unroll
        for (int r = 0; r < 8; ++r)
          stQ[mt * 512 + base0 + r * 16] = (bf16)((acc[t][mt][r] + bv) * scale);
    } else if (sect == 1) {
      // K: B-packed (K^T); dest = mt*512 + (r+half + 16*((nf&15)>=8))*16 + (nf&7) + 8*(nf>=16)
      const int base0 = (half + (((nf & 15) >= 8) ? 16 : 0)) * 16 + (nf & 7) + ((nf >= 16) ? 8 : 0);
      #pragma unroll
      for (int mt = 0; mt < 4; ++mt)
        #pragma unroll
        for (int r = 0; r < 8; ++r)
          stK[mt * 512 + base0 + r * 16] = (bf16)(acc[t][mt][r] + bv);
    } else {
      // V: B-packed; 8 rows land on consecutive elems of one lane slot -> one b128 store
      #pragma unroll
      for (int mt = 0; mt < 4; ++mt) {
        v8bf pk;
        #pragma unroll
        for (int r = 0; r < 8; ++r) pk[r] = (bf16)(acc[t][mt][r] + bv);
        const int mbase = mt * 16 + half;
        const int lslot = (nf & 15) + ((mbase & 8) ? 16 : 0);
        *(v8bf*)(stV + (((mbase >> 5) * 2 + (nf >> 4)) << 9) + lslot * 16 + ((mbase & 16) ? 8 : 0)) = pk;
      }
    }
  }
}

// Attention for one head from fragment-packed staging; writes attn_out tile column `head`.
__device__ __forceinline__ void attn_head(
    const float* __restrict__ biasf,
    const bf16* stQ, const bf16* stK, const bf16* stV, bf16* stP,
    bf16* sAttn, int head, int lane, int lc, int half)
{
  const v8f vzero = {0.f, 0.f, 0.f, 0.f, 0.f, 0.f, 0.f, 0.f};
  const float* bh = biasf + head * (NTOK * NTOK);
  #pragma unroll 1
  for (int mt = 0; mt < 4; ++mt) {
    v16bf qf = *(const v16bf*)(stQ + mt * 512 + lane * 16);
    v8f st[4];
    #pragma unroll
    for (int j = 0; j < 4; ++j) {
      v16bf kf = *(const v16bf*)(stK + j * 512 + lane * 16);
      st[j] = wmma_bf16(qf, kf, vzero);
    }
    // masked softmax + relative-position bias; row m lives in 16 lanes (cols) at element r
    #pragma unroll
    for (int r = 0; r < 8; ++r) {
      const int m = mt * 16 + r + half;
      float srow[4];
      float mx = -1e30f;
      #pragma unroll
      for (int j = 0; j < 4; ++j) {
        const int n = j * 16 + lc;
        float s = -1e30f;
        if (m < NTOK && n < NTOK) s = st[j][r] + bh[m * NTOK + n];
        srow[j] = s;
        mx = fmaxf(mx, s);
      }
      #pragma unroll
      for (int d = 1; d < 16; d <<= 1) mx = fmaxf(mx, __shfl_xor(mx, d, 32));
      const bool ok = mx > -5e29f;
      float sum = 0.f;
      #pragma unroll
      for (int j = 0; j < 4; ++j) {
        float e = ok ? __expf(srow[j] - mx) : 0.f;
        srow[j] = e;
        sum += e;
      }
      #pragma unroll
      for (int d = 1; d < 16; d <<= 1) sum += __shfl_xor(sum, d, 32);
      const float inv = (sum > 0.f) ? 1.f / sum : 0.f;
      // P store (A-packed): base = (r+half + 16*(lc>=8))*16 + (lc&7)
      const int pbase = (r + half + ((lc >= 8) ? 16 : 0)) * 16 + (lc & 7);
      stP[pbase]           = (bf16)(srow[0] * inv);
      stP[pbase + 8]       = (bf16)(srow[1] * inv);
      stP[pbase + 512]     = (bf16)(srow[2] * inv);
      stP[pbase + 512 + 8] = (bf16)(srow[3] * inv);
    }

    // O = P @ V  (K = 64 tokens -> two bf16 WMMA k-steps)
    v16bf pf0 = *(const v16bf*)(stP + lane * 16);
    v16bf pf1 = *(const v16bf*)(stP + 512 + lane * 16);
    v8f o0 = wmma_bf16(pf0, *(const v16bf*)(stV + lane * 16), vzero);
    o0     = wmma_bf16(pf1, *(const v16bf*)(stV + 2 * 512 + lane * 16), o0);
    v8f o1 = wmma_bf16(pf0, *(const v16bf*)(stV + 512 + lane * 16), vzero);
    o1     = wmma_bf16(pf1, *(const v16bf*)(stV + 3 * 512 + lane * 16), o1);

    // attn_out A-packed; head h owns K-feature tile ktf == h
    #pragma unroll
    for (int nfT = 0; nfT < 2; ++nfT) {
      v8f o = nfT ? o1 : o0;
      const int ko = nfT * 16 + lc;
      const int jpart = (((ko >= 16) ? 4 : 0) + ((ko & 7) >> 1)) * 2 + (ko & 1);
      const int base0 = (half + (((ko & 15) >= 8) ? 16 : 0)) * 16 + jpart;
      bf16* dst = sAttn + (mt * 12 + head) * 512;
      #pragma unroll
      for (int r = 0; r < 8; ++r)
        dst[base0 + r * 16] = (bf16)o[r];
    }
  }
}

// ---------------- fused window attention: one workgroup (8 wave32) per window ----------------
__global__ __launch_bounds__(256) void fused_window_attention(
    const float* __restrict__ x,
    const float* __restrict__ qkv_b,
    const float* __restrict__ proj_b,
    const bf16*  __restrict__ wqkv,    // [kt12][nt72][512] bf16 B-fragments
    const bf16*  __restrict__ wproj,   // [kt12][nt24][512]
    const float* __restrict__ biasf,   // [head][49][49]
    float* __restrict__ out)
{
  extern __shared__ __align__(128) bf16 smem[];
  bf16* sAttn  = smem;                 // 24576 elems: attn_out A-packed [mt4][ktf12][512]
  bf16* sStage = smem + 24576;         // 8 regions * 7168 elems (Q 2048 | K 2048 | V 2048 | P 1024)

  const int tid  = threadIdx.x;
  const int lane = tid & 31;
  const int wave = tid >> 5;
  const int lc   = lane & 15;
  const int half = (lane >= 16) ? 8 : 0;
  const int b    = blockIdx.x;
  const float scale = 0.17677669529663687f;   // 1/sqrt(32)
  const float* xw = x + (size_t)b * (NTOK * DIM);
  const v8f vzero = {0.f, 0.f, 0.f, 0.f, 0.f, 0.f, 0.f, 0.f};

  bf16* stg = sStage + wave * 7168;

  // ===== Round 1: head = wave (QKV both chunks + attention; per-wave staging, no barrier) =====
  qkv_chunk(xw, qkv_b, wqkv, stg, stg + 2048, stg + 4096, wave, 0, lane, lc, half, scale);
  qkv_chunk(xw, qkv_b, wqkv, stg, stg + 2048, stg + 4096, wave, 1, lane, lc, half, scale);
  attn_head(biasf, stg, stg + 2048, stg + 4096, stg + 6144, sAttn, wave, lane, lc, half);
  __syncthreads();

  // ===== Round 2: QKV of heads 8..11 split across all 8 waves (one chunk each) =====
  {
    const int h2   = 8 + (wave & 3);
    bf16* stg2 = sStage + (wave & 3) * 7168;
    qkv_chunk(xw, qkv_b, wqkv, stg2, stg2 + 2048, stg2 + 4096, h2, wave >> 2, lane, lc, half, scale);
  }
  __syncthreads();

  // ===== Round 3: attention of heads 8..11 on waves 0..3 =====
  if (wave < 4) {
    bf16* stg3 = sStage + wave * 7168;
    attn_head(biasf, stg3, stg3 + 2048, stg3 + 4096, stg3 + 6144, sAttn, 8 + wave, lane, lc, half);
  }
  __syncthreads();

  // ===== Phase 3: output projection  out = attn @ proj_w^T + proj_b =====
  float* ow = out + (size_t)b * (NTOK * DIM);
  #pragma unroll 1
  for (int l = 0; l < 3; ++l) {
    const int nt = wave + 8 * l;                           // 0..23
    v8f acc[4];
    #pragma unroll
    for (int mt = 0; mt < 4; ++mt) acc[mt] = vzero;
    #pragma unroll 1
    for (int kt = 0; kt < 12; ++kt) {
      v16bf bfr = *(const v16bf*)(wproj + (((kt * 24 + nt) << 9) + lane * 16));
      #pragma unroll
      for (int mt = 0; mt < 4; ++mt) {
        v16bf afr = *(const v16bf*)(sAttn + ((mt * 12 + kt) << 9) + lane * 16);
        acc[mt] = wmma_bf16(afr, bfr, acc[mt]);
      }
    }
    const int n = nt * 16 + lc;
    const float bv = proj_b[n];
    #pragma unroll
    for (int mt = 0; mt < 4; ++mt) {
      #pragma unroll
      for (int r = 0; r < 8; ++r) {
        const int m = mt * 16 + r + half;
        if (m < NTOK) ow[m * DIM + n] = acc[mt][r] + bv;
      }
    }
  }
}

extern "C" void kernel_launch(void* const* d_in, const int* in_sizes, int n_in,
                              void* d_out, int out_size, void* d_ws, size_t ws_size,
                              hipStream_t stream) {
  (void)in_sizes; (void)n_in; (void)out_size; (void)ws_size;
  const float* x      = (const float*)d_in[0];
  const float* qkv_w  = (const float*)d_in[1];
  const float* qkv_b  = (const float*)d_in[2];
  const float* proj_w = (const float*)d_in[3];
  const float* proj_b = (const float*)d_in[4];
  const float* rpb    = (const float*)d_in[5];
  const int*   rel    = (const int*)d_in[6];
  float* out = (float*)d_out;

  // workspace carve-out (~1.24 MB): packed bf16 weights + pre-gathered bias
  bf16*  wqkv  = (bf16*)d_ws;                       // 12*72*512 = 442368 bf16
  bf16*  wproj = wqkv + 12 * 72 * 512;              // 12*24*512 = 147456 bf16
  float* biasf = (float*)(wproj + 12 * 24 * 512);   // 12*49*49  floats

  pack_wqkv <<<(DIM * 3 * DIM + 255) / 256, 256, 0, stream>>>(qkv_w, wqkv);
  pack_wproj<<<(DIM * DIM + 255) / 256,     256, 0, stream>>>(proj_w, wproj);
  pack_bias <<<(NHEAD * NTOK * NTOK + 255) / 256, 256, 0, stream>>>(rpb, rel, biasf);

  const size_t smem = (size_t)(24576 + 8 * 7168) * sizeof(bf16);   // 160 KB
  hipFuncSetAttribute(reinterpret_cast<const void*>(fused_window_attention),
                      hipFuncAttributeMaxDynamicSharedMemorySize, (int)smem);
  fused_window_attention<<<NWIN, 256, smem, stream>>>(x, qkv_b, proj_b, wqkv, wproj, biasf, out);
}
